// MotionTrackingLK_335007450102
// MI455X (gfx1250) — compile-verified, matrix operand order checked by reading, not run on hardware
//
#include <hip/hip_runtime.h>
#include <hip/hip_bf16.h>
#include <math.h>

// MotionTrackingLK for MI455X (gfx1250, wave32).
// One wave per track; 8 waves (256 threads) per block; 256 blocks = 2048 tracks.
// Per frame pair: sample `first` once, Sobel once; per iteration: resample
// `second`, then compute the 2x3 Gram matrix (WP)^T P with
// v_wmma_f32_16x16x32_f16 over 14 K-chunks (K padded 441->448), f32 accumulate.
// Fragment operands are staged in LDS and loaded as two ds_load_b128 per
// operand; unused WMMA rows/cols read from a shared zero region (no branches).

typedef __attribute__((ext_vector_type(16))) _Float16 v16h;
typedef __attribute__((ext_vector_type(8)))  _Float16 v8h;
typedef __attribute__((ext_vector_type(8)))  float    v8f;

#define NWAVES 8
#define KPAD   448   // 441 window points padded to 14*32
#define NCHUNK 14

__device__ __forceinline__ float bilin512(const float* __restrict__ img,
                                          float px, float py) {
    // Matches reference: x = pos * (WIN*0.5); clip floor/floor+1 to [0,511];
    // weights computed from the clipped corner coordinates.
    float x = px * 10.5f, y = py * 10.5f;
    float xf = floorf(x), yf = floorf(y);
    float x0 = fminf(fmaxf(xf,        0.f), 511.f);
    float x1 = fminf(fmaxf(xf + 1.f,  0.f), 511.f);
    float y0 = fminf(fmaxf(yf,        0.f), 511.f);
    float y1 = fminf(fmaxf(yf + 1.f,  0.f), 511.f);
    float wa = (y1 - y) * (x1 - x);
    float wb = (y1 - y) * (x  - x0);
    float wc = (y  - y0) * (x1 - x);
    float wd = (y  - y0) * (x  - x0);
    int ix0 = (int)x0, ix1 = (int)x1, iy0 = (int)y0, iy1 = (int)y1;
    float Ia = img[iy0 * 512 + ix0];
    float Ib = img[iy0 * 512 + ix1];
    float Ic = img[iy1 * 512 + ix0];
    float Id = img[iy1 * 512 + ix1];
    return wa * Ia + wb * Ib + wc * Ic + wd * Id;
}

__global__ __launch_bounds__(256) void
MotionTrackingLK_kernel(const float* __restrict__ tracks,   // (8,256,2)
                        const float* __restrict__ imgs,     // (8,16,512,512)
                        float* __restrict__ out) {          // (8,256,16,2)
    // Per-wave staging in LDS (16B aligned for ds_load_b128 fragments).
    __shared__ __align__(16) float    sFirst[NWAVES][KPAD];  // first window (f32)
    __shared__ __align__(16) _Float16 sA[NWAVES][2 * KPAD];  // ch0=w*Ix ch1=w*Iy
    __shared__ __align__(16) _Float16 sB[NWAVES][3 * KPAD];  // ch0=Ix ch1=Iy ch2=b
    __shared__ __align__(16) _Float16 sZero[KPAD];           // shared zero rows

    const int lane = threadIdx.x & 31;
    const int wv   = threadIdx.x >> 5;
    const int g    = blockIdx.x * NWAVES + wv;        // track id 0..2047
    const int bb   = g >> 8;                          // batch
    const int nn   = g & 255;                         // track within batch
    const float cr = 512.0f / 21.0f;

    // Zero-init staging (padding 441..447 and the zero region stay 0 forever).
    for (int k = lane; k < KPAD;     k += 32) sFirst[wv][k] = 0.f;
    for (int k = lane; k < 2 * KPAD; k += 32) sA[wv][k] = (_Float16)0.f;
    for (int k = lane; k < 3 * KPAD; k += 32) sB[wv][k] = (_Float16)0.f;
    for (int k = threadIdx.x; k < KPAD; k += 256) sZero[k] = (_Float16)0.f;
    __syncthreads();

    float px = tracks[(bb * 256 + nn) * 2 + 0] * cr + cr;
    float py = tracks[(bb * 256 + nn) * 2 + 1] * cr + cr;

    // pos0 = raw track locations.
    if (lane == 0) {
        float* o = out + (size_t)((bb * 256 + nn) * 16) * 2;
        o[0] = (px - cr) / cr;
        o[1] = (py - cr) / cr;
    }

    const float* base = imgs + (size_t)bb * 16 * 512 * 512;
    float firstReg[NCHUNK];

    // WMMA operand base pointers (uniform per lane, chosen once):
    // lanes whose row/col is unused read the shared zero region.
    const int r  = lane & 15;     // A row / B col index
    const int hs = lane >> 4;     // half-wave select in the K packing
    const v8h* aFrag = (const v8h*)((r < 2) ? &sA[wv][r * KPAD] : sZero);
    const v8h* bFrag = (const v8h*)((r < 3) ? &sB[wv][r * KPAD] : sZero);

#pragma unroll 1
    for (int p = 0; p < 15; ++p) {                    // 15 frame pairs
        const float* f0 = base + (size_t)p * 512 * 512;
        const float* f1 = f0 + 512 * 512;

        // Warm L2 for the *next* pair's second frame near this track.
        if (p < 14) {
            const float* fn = f1 + 512 * 512;
            int cy = (int)fminf(fmaxf(py * 10.5f, 0.f), 511.f);
            int cx = (int)fminf(fmaxf(px * 10.5f, 0.f), 511.f);
            __builtin_prefetch(fn + (cy + (lane >> 1) - 8) * 512 + cx, 0, 1);
        }

        // ---- sample `first` at current sampler (once per pair) ----
#pragma unroll
        for (int t = 0; t < NCHUNK; ++t) {
            int k = t * 32 + lane;
            float v = 0.f;
            if (k < 441) {
                int i = k / 21, j = k - i * 21;
                v = bilin512(f0, px + (-1.0f + 0.1f * j),
                                 py + (-1.0f + 0.1f * i));
            }
            firstReg[t] = v;
            sFirst[wv][k] = v;
        }
        __syncthreads();

        // ---- Sobel gradients + Gaussian weights (constant per pair) ----
#pragma unroll
        for (int t = 0; t < NCHUNK; ++t) {
            int k = t * 32 + lane;
            if (k < 441) {
                int i = k / 21, j = k - i * 21;
                const float* F = sFirst[wv];
                auto at = [&](int ii, int jj) -> float {
                    return (ii >= 0 && ii < 21 && jj >= 0 && jj < 21)
                               ? F[ii * 21 + jj] : 0.f;     // SAME zero pad
                };
                float Ix = -at(i-1,j-1) + at(i-1,j+1)
                           - 2.f*at(i,j-1) + 2.f*at(i,j+1)
                           - at(i+1,j-1) + at(i+1,j+1);
                float Iy = -at(i-1,j-1) - 2.f*at(i-1,j) - at(i-1,j+1)
                           + at(i+1,j-1) + 2.f*at(i+1,j) + at(i+1,j+1);
                float dx = (float)(j - 10), dy = (float)(i - 10);
                float w = 0.19947114020071635f * __expf(-0.125f * (dx*dx + dy*dy));
                sA[wv][0 * KPAD + k] = (_Float16)(w * Ix);
                sA[wv][1 * KPAD + k] = (_Float16)(w * Iy);
                sB[wv][0 * KPAD + k] = (_Float16)Ix;
                sB[wv][1 * KPAD + k] = (_Float16)Iy;
            }
        }
        __syncthreads();

#pragma unroll 1
        for (int it = 0; it < 5; ++it) {              // Gauss-Newton iterations
            // ---- resample `second`, b = first - second = -(second-first) ----
#pragma unroll
            for (int t = 0; t < NCHUNK; ++t) {
                int k = t * 32 + lane;
                if (k < 441) {
                    int i = k / 21, j = k - i * 21;
                    float sec = bilin512(f1, px + (-1.0f + 0.1f * j),
                                             py + (-1.0f + 0.1f * i));
                    sB[wv][2 * KPAD + k] = (_Float16)(firstReg[t] - sec);
                }
            }
            __syncthreads();

            // ---- Gram matrix (WP)^T P via v_wmma_f32_16x16x32_f16 ----
            // Per lane, the 16-bit A/B K-packing is two contiguous 8-half runs:
            // [kb, kb+8) and [kb+16, kb+24), kb = 32c + 8*hs -> 2x ds_load_b128.
            v8f acc = {};
#pragma unroll 2
            for (int c = 0; c < NCHUNK; ++c) {
                int q = c * 4 + hs;                   // (32c + 8hs) / 8
                v8h aLo = aFrag[q],     bLo = bFrag[q];
                v8h aHi = aFrag[q + 2], bHi = bFrag[q + 2];
                v16h af = __builtin_shufflevector(aLo, aHi,
                              0,1,2,3,4,5,6,7,8,9,10,11,12,13,14,15);
                v16h bf = __builtin_shufflevector(bLo, bHi,
                              0,1,2,3,4,5,6,7,8,9,10,11,12,13,14,15);
                acc = __builtin_amdgcn_wmma_f32_16x16x32_f16(
                          false, af, false, bf, (short)0, acc, false, false);
            }

            // ---- extract D entries (C/D layout: lanes 0..15, vgpr m = row m) ----
            float a00 = __shfl(acc[0], 0, 32);   // D[0,0] = sum w*Ix*Ix
            float a01 = __shfl(acc[0], 1, 32);   // D[0,1] = sum w*Ix*Iy
            float b0  = __shfl(acc[0], 2, 32);   // D[0,2] = ATb_x
            float a11 = __shfl(acc[1], 1, 32);   // D[1,1] = sum w*Iy*Iy
            float b1  = __shfl(acc[1], 2, 32);   // D[1,2] = ATb_y

            float invdet = 1.0f / (a00 * a11 - a01 * a01 + 1e-7f);
            float vx = invdet * ( a11 * b0 - a01 * b1);
            float vy = invdet * (-a01 * b0 + a00 * b1);
            px += vx;
            py += vy;
            __syncthreads();
        }

        // pos_{p+1}
        if (lane == 0) {
            float* o = out + (size_t)(((bb * 256 + nn) * 16) + (p + 1)) * 2;
            o[0] = (px - cr) / cr;
            o[1] = (py - cr) / cr;
        }
    }
}

extern "C" void kernel_launch(void* const* d_in, const int* in_sizes, int n_in,
                              void* d_out, int out_size, void* d_ws, size_t ws_size,
                              hipStream_t stream) {
    const float* tracks = (const float*)d_in[0];   // (8,256,2) f32
    const float* imgs   = (const float*)d_in[1];   // (8,16,512,512,1) f32
    float* out          = (float*)d_out;           // (8,256,16,2) f32
    (void)in_sizes; (void)n_in; (void)out_size; (void)d_ws; (void)ws_size;

    dim3 grid(256);    // 256 blocks * 8 waves = 2048 tracks
    dim3 block(256);
    MotionTrackingLK_kernel<<<grid, block, 0, stream>>>(tracks, imgs, out);
}
